// TransformerEncoderLayer_40785009443433
// MI455X (gfx1250) — matrix operand order repaired and checked by reading, not compiled, not run on hardware
//
#include <hip/hip_runtime.h>
#include <hip/hip_bf16.h>

// ---------------------------------------------------------------------------
// Types for gfx1250 WMMA
// ---------------------------------------------------------------------------
typedef __bf16 bf16;
typedef __attribute__((ext_vector_type(16))) __bf16 v16bf;
typedef __attribute__((ext_vector_type(8)))  float  v8f;

union F16U { v16bf v; uint4 u[2]; };

__device__ __forceinline__ v8f wmma_bf16(v16bf a, v16bf b, v8f c) {
  // v_wmma_f32_16x16x32_bf16  (D = A x B + C, fp32 accumulate)
  return __builtin_amdgcn_wmma_f32_16x16x32_bf16(false, a, false, b, (short)0, c, false, false);
}

#define TB  2
#define TS  2048
#define TD  1024
#define THN 16
#define THD 64
#define TF  4096
#define TM  (TB * TS)   /* 4096 rows */

// ---------------------------------------------------------------------------
// Weight convert + transpose: W[K][N] fp32 -> Wt[N][K] bf16
// ---------------------------------------------------------------------------
__global__ __launch_bounds__(256)
void convert_wt_kernel(const float* __restrict__ W, bf16* __restrict__ Wt, int K, int N) {
  size_t i = (size_t)blockIdx.x * 256 + threadIdx.x;   // grid sized exactly K*N/256
  int k = (int)(i / (size_t)N);
  int n = (int)(i % (size_t)N);
  Wt[(size_t)n * K + k] = (bf16)W[i];
}

// ---------------------------------------------------------------------------
// V[4096][1024] bf16 (head-major cols) -> Vt[B][H][HD][S] bf16
// ---------------------------------------------------------------------------
__global__ __launch_bounds__(256)
void transpose_v_kernel(const bf16* __restrict__ V, bf16* __restrict__ Vt) {
  int i  = blockIdx.x * 256 + threadIdx.x;  // 4M elements, write-coalesced
  int s  = i & (TS - 1);
  int d  = (i >> 11) & (THD - 1);
  int bh = i >> 17;                 // b*16 + h
  int b  = bh >> 4, h = bh & 15;
  Vt[i] = V[(size_t)(b * TS + s) * TD + h * THD + d];
}

// ---------------------------------------------------------------------------
// LayerNorm (fp32 in) -> bf16 out.  One 256-thread block per row of 1024.
// ---------------------------------------------------------------------------
__global__ __launch_bounds__(256)
void layernorm_bf16_kernel(const float* __restrict__ X, const float* __restrict__ g,
                           const float* __restrict__ bta, bf16* __restrict__ Y) {
  __shared__ float red[16];
  const int t = threadIdx.x;
  const float* xr = X + (size_t)blockIdx.x * TD;
  float4 v  = *(const float4*)&xr[t * 4];
  float s   = v.x + v.y + v.z + v.w;
  float s2  = v.x * v.x + v.y * v.y + v.z * v.z + v.w * v.w;
  #pragma unroll
  for (int m = 16; m >= 1; m >>= 1) {
    s  += __shfl_xor(s,  m, 32);
    s2 += __shfl_xor(s2, m, 32);
  }
  if ((t & 31) == 0) { red[(t >> 5) * 2] = s; red[(t >> 5) * 2 + 1] = s2; }
  __syncthreads();
  float ts = 0.f, ts2 = 0.f;
  #pragma unroll
  for (int w = 0; w < 8; ++w) { ts += red[2 * w]; ts2 += red[2 * w + 1]; }
  float mu   = ts * (1.0f / TD);
  float var  = ts2 * (1.0f / TD) - mu * mu;
  float rstd = rsqrtf(var + 1e-5f);
  float4 gv = *(const float4*)&g[t * 4];
  float4 bv = *(const float4*)&bta[t * 4];
  bf16* yr = Y + (size_t)blockIdx.x * TD;
  yr[t * 4 + 0] = (bf16)((v.x - mu) * rstd * gv.x + bv.x);
  yr[t * 4 + 1] = (bf16)((v.y - mu) * rstd * gv.y + bv.y);
  yr[t * 4 + 2] = (bf16)((v.z - mu) * rstd * gv.z + bv.z);
  yr[t * 4 + 3] = (bf16)((v.w - mu) * rstd * gv.w + bv.w);
}

// ---------------------------------------------------------------------------
// WMMA GEMM: C[M][N] = A[M][K] @ B[K][N],  A bf16 row-major, Bt bf16 [N][K].
// Block tile 128x128, K-step 32, double-buffered LDS (one barrier / K-step).
// 8 waves; each wave computes 64x32 via 8 x v_wmma_f32_16x16x32_bf16.
// EPI: 0 = bias -> bf16 ; 1 = bias + exact GELU -> bf16 ; 2 = bias + residual -> fp32
// ---------------------------------------------------------------------------
template <int EPI>
__global__ __launch_bounds__(256)
void gemm_bf16_kernel(const bf16* __restrict__ A, const bf16* __restrict__ Bt,
                      const float* __restrict__ bias, const float* __restrict__ res,
                      void* __restrict__ out, int M, int N, int K) {
  constexpr int PA = 48;                 // LDS row pitch (elems): 96B, 16B aligned rows
  __shared__ bf16 sA[2][128 * PA];       // 2 x 12 KB
  __shared__ bf16 sB[2][128 * PA];       // 2 x 12 KB

  const int t    = threadIdx.x;
  const int wid  = t >> 5;
  const int lane = t & 31;
  const int l16  = lane & 15;
  const int lh   = lane >> 4;
  const int wm   = wid >> 2;             // 0..1  (64 rows each)
  const int wn   = wid & 3;              // 0..3  (32 cols each)
  const int rowBase = blockIdx.y * 128;
  const int colBase = blockIdx.x * 128;

  // per-thread tile-copy coordinates (two passes of 256 b128 vectors)
  const int ra = t >> 2;                 // 0..63
  const int ca = (t & 3) * 8;            // 0,8,16,24
  const size_t aOff0 = (size_t)(rowBase + ra) * K + ca;
  const size_t aOff1 = (size_t)(rowBase + ra + 64) * K + ca;
  const size_t bOff0 = (size_t)(colBase + ra) * K + ca;
  const size_t bOff1 = (size_t)(colBase + ra + 64) * K + ca;
  const int lA0 = ra * PA + ca;
  const int lA1 = (ra + 64) * PA + ca;

  v8f acc[4][2];
  v8f zero = {};
  #pragma unroll
  for (int i = 0; i < 4; ++i)
    #pragma unroll
    for (int j = 0; j < 2; ++j) acc[i][j] = zero;

  // prologue: tile 0 -> regs -> LDS buffer 0
  uint4 rA0 = *(const uint4*)&A[aOff0];
  uint4 rA1 = *(const uint4*)&A[aOff1];
  uint4 rB0 = *(const uint4*)&Bt[bOff0];
  uint4 rB1 = *(const uint4*)&Bt[bOff1];
  *(uint4*)&sA[0][lA0] = rA0;
  *(uint4*)&sA[0][lA1] = rA1;
  *(uint4*)&sB[0][lA0] = rB0;
  *(uint4*)&sB[0][lA1] = rB1;

  int cur = 0;
  for (int k0 = 0; k0 < K; k0 += 32) {
    __syncthreads();                     // buffer `cur` ready; prior readers of cur^1 done
    const bool hasNext = (k0 + 32 < K);
    if (hasNext) {                       // issue next tile loads; latency hides behind WMMA
      rA0 = *(const uint4*)&A[aOff0 + k0 + 32];
      rA1 = *(const uint4*)&A[aOff1 + k0 + 32];
      rB0 = *(const uint4*)&Bt[bOff0 + k0 + 32];
      rB1 = *(const uint4*)&Bt[bOff1 + k0 + 32];
    }

    const bf16* cA = sA[cur];
    const bf16* cB = sB[cur];
    v16bf af[4], bfr[2];
    #pragma unroll
    for (int i = 0; i < 4; ++i) {        // A frag: lane = M row
      int row = wm * 64 + i * 16 + l16;
      F16U fu;
      fu.u[0] = *(const uint4*)&cA[row * PA + lh * 8];
      fu.u[1] = *(const uint4*)&cA[row * PA + 16 + lh * 8];
      af[i] = fu.v;
    }
    #pragma unroll
    for (int j = 0; j < 2; ++j) {        // B frag: lane = N col (Bt row)
      int row = wn * 32 + j * 16 + l16;
      F16U fu;
      fu.u[0] = *(const uint4*)&cB[row * PA + lh * 8];
      fu.u[1] = *(const uint4*)&cB[row * PA + 16 + lh * 8];
      bfr[j] = fu.v;
    }
    #pragma unroll
    for (int i = 0; i < 4; ++i)
      #pragma unroll
      for (int j = 0; j < 2; ++j)
        acc[i][j] = wmma_bf16(af[i], bfr[j], acc[i][j]);

    if (hasNext) {                       // stash next tile in the other buffer
      bf16* nA = sA[cur ^ 1];
      bf16* nB = sB[cur ^ 1];
      *(uint4*)&nA[lA0] = rA0;
      *(uint4*)&nA[lA1] = rA1;
      *(uint4*)&nB[lA0] = rB0;
      *(uint4*)&nB[lA1] = rB1;
    }
    cur ^= 1;
  }

  // Epilogue.  C layout: lane holds col n=l16; vgpr r holds row m = 8*lh + r.
  #pragma unroll
  for (int i = 0; i < 4; ++i) {
    int row0 = rowBase + wm * 64 + i * 16 + lh * 8;
    #pragma unroll
    for (int j = 0; j < 2; ++j) {
      int col = colBase + wn * 32 + j * 16 + l16;
      float bv = bias[col];
      #pragma unroll
      for (int r = 0; r < 8; ++r) {
        size_t idx = (size_t)(row0 + r) * N + col;
        float x = acc[i][j][r] + bv;
        if (EPI == 0) {
          ((bf16*)out)[idx] = (bf16)x;
        } else if (EPI == 1) {
          float gel = 0.5f * x * (1.0f + erff(x * 0.70710678118654752f));
          ((bf16*)out)[idx] = (bf16)gel;
        } else {
          ((float*)out)[idx] = x + res[idx];
        }
      }
    }
  }
}

// ---------------------------------------------------------------------------
// Flash attention.  Grid (S/128 q-tiles, H, B); 256 threads = 8 waves.
// Wave w owns q rows [128*qt + 16*w, +16).  64-key tiles; online softmax.
// K/V tiles double-buffered; buffer 1 aliases the (dead) Q staging tile.
// LDS map (bytes): [0,18432) Q tile, later K/V buffer 1 (K@0, V@9216);
//                  [18432,27648) K buffer 0; [27648,36864) V buffer 0;
//                  [36864,55296) per-wave P staging.
// Buffer base selected by runtime offset arithmetic (no LDS pointer arrays —
// constant-initialized pointer aggregates fail to link).
// ---------------------------------------------------------------------------
__global__ __launch_bounds__(256)
void attention_kernel(const bf16* __restrict__ Q, const bf16* __restrict__ Kb,
                      const bf16* __restrict__ Vt, const unsigned char* __restrict__ mask,
                      bf16* __restrict__ Ctx) {
  constexpr int PQ = 72;                 // pitch for [*][64] tiles (144B rows, 16B aligned)
  __shared__ __align__(16) char smem[55296];
  char* smb = smem;
  bf16* sQ = (bf16*)smb;                 // 18432 B (dead after aq frags -> reused below)
  bf16* sP = (bf16*)(smb + 36864);       // 18432 B, wave-private P staging

  const int t    = threadIdx.x;
  const int wid  = t >> 5;
  const int lane = t & 31;
  const int l16  = lane & 15;
  const int lh   = lane >> 4;
  const int qt = blockIdx.x, h = blockIdx.y, b = blockIdx.z;
  const size_t rowQ0 = (size_t)b * TS + qt * 128;
  const int colH = h * THD;
  const size_t vtBase = (size_t)(b * THN + h) * THD * TS;

  // Load 128x64 Q tile
  #pragma unroll
  for (int p = 0; p < 4; ++p) {
    int vecId = p * 256 + t;             // 1024 vecs, 8 per row
    int r = vecId >> 3, c = (vecId & 7) * 8;
    *(uint4*)&sQ[r * PQ + c] = *(const uint4*)&Q[(rowQ0 + r) * TD + colH + c];
  }
  __syncthreads();

  v16bf aq[2];                           // Q A-fragments, resident for whole kernel
  #pragma unroll
  for (int kk = 0; kk < 2; ++kk) {
    F16U fu;
    int row = wid * 16 + l16;
    fu.u[0] = *(const uint4*)&sQ[row * PQ + kk * 32 + lh * 8];
    fu.u[1] = *(const uint4*)&sQ[row * PQ + kk * 32 + 16 + lh * 8];
    aq[kk] = fu.v;
  }
  // NOTE: __syncthreads() at loop top waits dscnt==0, so every wave's aq reads
  // complete before any wave first stores into buffer 1 (alias of sQ).

  v8f zero = {};
  float mrow[8], lrow[8];
  v8f accO[4];
  #pragma unroll
  for (int r = 0; r < 8; ++r) { mrow[r] = -1e30f; lrow[r] = 0.f; }
  #pragma unroll
  for (int n = 0; n < 4; ++n) accO[n] = zero;

  bf16* pw = &sP[wid * 16 * PQ];

  // per-thread K/V tile-copy coordinates (two passes of 256 b128 vectors)
  const int rr = t >> 3;                 // 0..31
  const int cc = (t & 7) * 8;
  const size_t kOff0 = ((size_t)b * TS + rr) * TD + colH + cc;        // + s0*TD per tile
  const size_t kOff1 = ((size_t)b * TS + 32 + rr) * TD + colH + cc;
  const size_t vOff0 = vtBase + (size_t)rr * TS + cc;                 // + s0 per tile
  const size_t vOff1 = vtBase + (size_t)(32 + rr) * TS + cc;
  const int lK0 = rr * PQ + cc;
  const int lK1 = (32 + rr) * PQ + cc;

  // prologue: key-tile 0 -> regs -> buffer 0
  uint4 rK0 = *(const uint4*)&Kb[kOff0];
  uint4 rK1 = *(const uint4*)&Kb[kOff1];
  uint4 rV0 = *(const uint4*)&Vt[vOff0];
  uint4 rV1 = *(const uint4*)&Vt[vOff1];
  {
    bf16* sK0p = (bf16*)(smb + 18432);
    bf16* sV0p = (bf16*)(smb + 27648);
    *(uint4*)&sK0p[lK0] = rK0;
    *(uint4*)&sK0p[lK1] = rK1;
    *(uint4*)&sV0p[lK0] = rV0;
    *(uint4*)&sV0p[lK1] = rV1;
  }

  int cur = 0;
  for (int kt = 0; kt < TS / 64; ++kt) {
    const int s0 = kt * 64;
    __syncthreads();                     // buffer `cur` ready; readers of cur^1 done
    const bool hasNext = (kt + 1 < TS / 64);
    if (hasNext) {                       // issue next tile loads now; hide behind compute
      rK0 = *(const uint4*)&Kb[kOff0 + (size_t)(s0 + 64) * TD];
      rK1 = *(const uint4*)&Kb[kOff1 + (size_t)(s0 + 64) * TD];
      rV0 = *(const uint4*)&Vt[vOff0 + s0 + 64];
      rV1 = *(const uint4*)&Vt[vOff1 + s0 + 64];
    }
    // runtime-selected double-buffer bases: buf0 -> K@18432/V@27648, buf1 -> K@0/V@9216
    const int base = cur ? 0 : 18432;    // K base for current buffer
    const bf16* sK = (const bf16*)(smb + base);
    const bf16* sV = (const bf16*)(smb + base + 9216);

    // S = Q @ K^T (per-wave 16q x 64k), scaled and masked
    v8f sacc[4];
    #pragma unroll
    for (int jn = 0; jn < 4; ++jn) {
      v8f a = zero;
      #pragma unroll
      for (int kk = 0; kk < 2; ++kk) {
        F16U fb;
        int row = jn * 16 + l16;         // key row = B column
        fb.u[0] = *(const uint4*)&sK[row * PQ + kk * 32 + lh * 8];
        fb.u[1] = *(const uint4*)&sK[row * PQ + kk * 32 + 16 + lh * 8];
        a = wmma_bf16(aq[kk], fb.v, a);
      }
      int skey = s0 + jn * 16 + l16;
      unsigned char mv = mask[b * TS + skey];
      #pragma unroll
      for (int r = 0; r < 8; ++r) {
        float x = a[r] * 0.125f;         // 1/sqrt(64)
        a[r] = mv ? x : -1e30f;
      }
      sacc[jn] = a;
    }

    // online softmax: row max / sum across 16 lanes of each half (fixed rows)
    float cmax[8], csum[8], mnew[8], alpha[8];
    #pragma unroll
    for (int r = 0; r < 8; ++r) {
      float mx = sacc[0][r];
      #pragma unroll
      for (int jn = 1; jn < 4; ++jn) mx = fmaxf(mx, sacc[jn][r]);
      cmax[r] = mx;
    }
    #pragma unroll
    for (int m = 1; m < 16; m <<= 1)
      #pragma unroll
      for (int r = 0; r < 8; ++r)
        cmax[r] = fmaxf(cmax[r], __shfl_xor(cmax[r], m, 32));
    #pragma unroll
    for (int r = 0; r < 8; ++r) {
      mnew[r]  = fmaxf(mrow[r], cmax[r]);
      alpha[r] = __expf(mrow[r] - mnew[r]);
      mrow[r]  = mnew[r];
      csum[r]  = 0.f;
    }

    // P = exp(S - m), staged through wave-private LDS for C-layout -> A-layout
    #pragma unroll
    for (int jn = 0; jn < 4; ++jn)
      #pragma unroll
      for (int r = 0; r < 8; ++r) {
        float p = __expf(sacc[jn][r] - mnew[r]);
        csum[r] += p;
        pw[(lh * 8 + r) * PQ + jn * 16 + l16] = (bf16)p;
      }
    #pragma unroll
    for (int m = 1; m < 16; m <<= 1)
      #pragma unroll
      for (int r = 0; r < 8; ++r)
        csum[r] += __shfl_xor(csum[r], m, 32);
    #pragma unroll
    for (int r = 0; r < 8; ++r) lrow[r] = lrow[r] * alpha[r] + csum[r];
    #pragma unroll
    for (int n = 0; n < 4; ++n)
      #pragma unroll
      for (int r = 0; r < 8; ++r)
        accO[n][r] *= alpha[r];

    // O += P @ V   (K-dim = 64 keys -> two WMMA k-steps)
    #pragma unroll
    for (int kk = 0; kk < 2; ++kk) {
      F16U fa;
      fa.u[0] = *(const uint4*)&pw[l16 * PQ + kk * 32 + lh * 8];
      fa.u[1] = *(const uint4*)&pw[l16 * PQ + kk * 32 + 16 + lh * 8];
      #pragma unroll
      for (int n = 0; n < 4; ++n) {
        F16U fb;
        int rowv = n * 16 + l16;         // d row of Vt tile = B column
        fb.u[0] = *(const uint4*)&sV[rowv * PQ + kk * 32 + lh * 8];
        fb.u[1] = *(const uint4*)&sV[rowv * PQ + kk * 32 + 16 + lh * 8];
        accO[n] = wmma_bf16(fa.v, fb.v, accO[n]);
      }
    }

    if (hasNext) {                       // stash next K/V tile in the other buffer
      const int nbase = cur ? 18432 : 0;
      bf16* sKn = (bf16*)(smb + nbase);
      bf16* sVn = (bf16*)(smb + nbase + 9216);
      *(uint4*)&sKn[lK0] = rK0;
      *(uint4*)&sKn[lK1] = rK1;
      *(uint4*)&sVn[lK0] = rV0;
      *(uint4*)&sVn[lK1] = rV1;
    }
    cur ^= 1;
  }

  // normalize and store context (bf16, head-major columns)
  #pragma unroll
  for (int r = 0; r < 8; ++r) lrow[r] = 1.0f / lrow[r];
  size_t qrow0 = rowQ0 + wid * 16 + lh * 8;
  #pragma unroll
  for (int n = 0; n < 4; ++n) {
    int col = colH + n * 16 + l16;
    #pragma unroll
    for (int r = 0; r < 8; ++r)
      Ctx[(qrow0 + r) * TD + col] = (bf16)(accO[n][r] * lrow[r]);
  }
}

// ---------------------------------------------------------------------------
// Launch
// ---------------------------------------------------------------------------
extern "C" void kernel_launch(void* const* d_in, const int* in_sizes, int n_in,
                              void* d_out, int out_size, void* d_ws, size_t ws_size,
                              hipStream_t stream) {
  (void)in_sizes; (void)n_in; (void)out_size; (void)ws_size;
  const float* feature = (const float*)d_in[0];
  const unsigned char* mask = (const unsigned char*)d_in[1];
  const float* wq  = (const float*)d_in[2];
  const float* pbq = (const float*)d_in[3];
  const float* wk  = (const float*)d_in[4];
  const float* pbk = (const float*)d_in[5];
  const float* wv  = (const float*)d_in[6];
  const float* pbv = (const float*)d_in[7];
  const float* wo  = (const float*)d_in[8];
  const float* pbo = (const float*)d_in[9];
  const float* ln1g = (const float*)d_in[10];
  const float* ln1b = (const float*)d_in[11];
  const float* ln2g = (const float*)d_in[12];
  const float* ln2b = (const float*)d_in[13];
  const float* w1  = (const float*)d_in[14];
  const float* pb1 = (const float*)d_in[15];
  const float* w2  = (const float*)d_in[16];
  const float* pb2 = (const float*)d_in[17];

  char* ws = (char*)d_ws;
  const size_t MB = 1ull << 20;
  bf16* Wqt = (bf16*)(ws + 0 * MB);     // 2 MB each for DxD
  bf16* Wkt = (bf16*)(ws + 2 * MB);
  bf16* Wvt = (bf16*)(ws + 4 * MB);
  bf16* Wot = (bf16*)(ws + 6 * MB);
  bf16* W1t = (bf16*)(ws + 8 * MB);     // 8 MB
  bf16* W2t = (bf16*)(ws + 16 * MB);    // 8 MB
  bf16* Xn  = (bf16*)(ws + 24 * MB);    // 8 MB  (LN output, reused for LN2)
  bf16* Qb  = (bf16*)(ws + 32 * MB);    // 8 MB
  bf16* Kb  = (bf16*)(ws + 40 * MB);    // 8 MB
  bf16* Vb  = (bf16*)(ws + 48 * MB);    // 8 MB
  bf16* Vtb = (bf16*)(ws + 56 * MB);    // 8 MB
  bf16* Ctx = (bf16*)(ws + 64 * MB);    // 8 MB
  bf16* F1  = (bf16*)(ws + 72 * MB);    // 32 MB (GELU output 4096x4096)
  float* AttnRes = (float*)(ws + 104 * MB); // 16 MB fp32

  // 1) weights -> bf16 [N][K]
  convert_wt_kernel<<<(TD * TD) / 256, 256, 0, stream>>>(wq, Wqt, TD, TD);
  convert_wt_kernel<<<(TD * TD) / 256, 256, 0, stream>>>(wk, Wkt, TD, TD);
  convert_wt_kernel<<<(TD * TD) / 256, 256, 0, stream>>>(wv, Wvt, TD, TD);
  convert_wt_kernel<<<(TD * TD) / 256, 256, 0, stream>>>(wo, Wot, TD, TD);
  convert_wt_kernel<<<(TD * TF) / 256, 256, 0, stream>>>(w1, W1t, TD, TF);
  convert_wt_kernel<<<(TF * TD) / 256, 256, 0, stream>>>(w2, W2t, TF, TD);

  // 2) LN1
  layernorm_bf16_kernel<<<TM, 256, 0, stream>>>(feature, ln1g, ln1b, Xn);

  // 3) Q/K/V projections
  dim3 gDD(TD / 128, TM / 128);
  gemm_bf16_kernel<0><<<gDD, 256, 0, stream>>>(Xn, Wqt, pbq, nullptr, Qb, TM, TD, TD);
  gemm_bf16_kernel<0><<<gDD, 256, 0, stream>>>(Xn, Wkt, pbk, nullptr, Kb, TM, TD, TD);
  gemm_bf16_kernel<0><<<gDD, 256, 0, stream>>>(Xn, Wvt, pbv, nullptr, Vb, TM, TD, TD);
  transpose_v_kernel<<<(TM * TD) / 256, 256, 0, stream>>>(Vb, Vtb);

  // 4) flash attention
  dim3 gAtt(TS / 128, THN, TB);
  attention_kernel<<<gAtt, 256, 0, stream>>>(Qb, Kb, Vtb, mask, Ctx);

  // 5) out projection + residual (fp32)
  gemm_bf16_kernel<2><<<gDD, 256, 0, stream>>>(Ctx, Wot, pbo, feature, AttnRes, TM, TD, TD);

  // 6) LN2 + FFN
  layernorm_bf16_kernel<<<TM, 256, 0, stream>>>(AttnRes, ln2g, ln2b, Xn);
  dim3 gDF(TF / 128, TM / 128);
  gemm_bf16_kernel<1><<<gDF, 256, 0, stream>>>(Xn, W1t, pb1, nullptr, F1, TM, TF, TD);
  dim3 gFD(TD / 128, TM / 128);
  gemm_bf16_kernel<2><<<gFD, 256, 0, stream>>>(F1, W2t, pb2, AttnRes, (float*)d_out, TM, TD, TF);
}
